// LSTMQNet_53017076301971
// MI455X (gfx1250) — matrix operand order repaired and checked by reading, not compiled, not run on hardware
//
#include <hip/hip_runtime.h>
#include <hip/hip_bf16.h>
#include <math.h>

// ---------------------------------------------------------------- types
typedef _Float16 half_t;
typedef __attribute__((ext_vector_type(16))) _Float16 v16h;
typedef __attribute__((ext_vector_type(8)))  _Float16 v8h;
typedef __attribute__((ext_vector_type(8)))  float    v8f;
typedef unsigned int       u32;
typedef unsigned long long u64;

// ---------------------------------------------------------------- problem dims
#define BB   256
#define LL   128
#define HH   8
#define FEAT 182
#define HID  512
#define NROW (BB*LL*HH)        // 262144 encoder rows
#define NBL  (BB*LL)           // 32768  (b,l) rows

// workspace offsets (in half_t units)
#define HO_WENC 0u                          // 11 * 128 * 192
#define HO_WIH  270336u                     // 2048*512
#define HO_WHH  1318912u
#define HO_WS1  2367488u                    // 512*512
#define HO_WA1  2629632u
#define HO_H0   2891776u                    // 32768*512
#define HO_ENC  19668992u                   // 262144*64 == 32768*512
#define HO_HN   36446208u                   // 32768*512

struct EncW { const float* W[11]; };
struct EncB { const float* b[11]; };

// ---------------------------------------------------------------- wmma helpers
static __device__ inline v8f wmma16(v16h a, v16h b, v8f c) {
    return __builtin_amdgcn_wmma_f32_16x16x32_f16(false, a, false, b,
                                                  (short)0, c, false, false);
}

// A fragment: row-major M x K tile, lane holds row m = m0+(lane&15),
// K = k0 + (lane<16 ? 0..15 : 16..31) contiguous
static __device__ inline v16h ld_A(const half_t* base, int ld, int m0, int k0) {
    int lane = threadIdx.x & 31;
    const half_t* p = base + (size_t)(m0 + (lane & 15)) * ld + k0 + ((lane >> 4) << 4);
    v8h lo = *(const v8h*)p;
    v8h hi = *(const v8h*)(p + 8);
    return __builtin_shufflevector(lo, hi, 0,1,2,3,4,5,6,7,8,9,10,11,12,13,14,15);
}

// B fragment from row-major weights W[n][k] (B[k][n] = W[n][k]):
// lane holds col n = n0+(lane&15), K = k0 + (lane<16 ? 0..15 : 16..31) contiguous
static __device__ inline v16h ld_B(const half_t* w, int ld, int n0, int k0) {
    int lane = threadIdx.x & 31;
    const half_t* p = w + (size_t)(n0 + (lane & 15)) * ld + k0 + ((lane >> 4) << 4);
    return *(const v16h*)p;
}

// async global->LDS copy of one 16B chunk per lane (ASYNCcnt-tracked)
static __device__ inline void async_cp16(u32 lds_byte_addr, u64 gaddr) {
    asm volatile("global_load_async_to_lds_b128 %0, %1, off"
                 :: "v"(lds_byte_addr), "v"(gaddr) : "memory");
}
static __device__ inline void async_wait0() {
    asm volatile("s_wait_asynccnt 0x0" ::: "memory");
}

static __device__ inline float sigm(float x) { return 1.0f / (1.0f + __expf(-x)); }

// ---------------------------------------------------------------- 0) weight pack
__global__ __launch_bounds__(256) void cvt_kernel(
    EncW ew, const float* W_ih, const float* W_hh, const float* Ws1,
    const float* Wa1, const float* hns, half_t* ws)
{
    size_t tid = (size_t)blockIdx.x * blockDim.x + threadIdx.x;
    size_t nt  = (size_t)gridDim.x * blockDim.x;

    half_t* wEnc = ws + HO_WENC;
    for (size_t i = tid; i < 11u * 128u * 192u; i += nt) {
        int layer = (int)(i / (128 * 192));
        int rc    = (int)(i % (128 * 192));
        int r = rc / 192, c = rc % 192;
        int out_d = (layer < 10) ? 128 : 64;
        int in_d  = (layer == 0) ? FEAT : 128;
        float v = 0.0f;
        if (r < out_d && c < in_d) v = ew.W[layer][r * in_d + c];
        wEnc[i] = (half_t)v;
    }
    half_t* wIh = ws + HO_WIH;
    half_t* wHh = ws + HO_WHH;
    for (size_t i = tid; i < 2048u * 512u; i += nt) {
        wIh[i] = (half_t)W_ih[i];
        wHh[i] = (half_t)W_hh[i];
    }
    half_t* wS1 = ws + HO_WS1;
    half_t* wA1 = ws + HO_WA1;
    for (size_t i = tid; i < 512u * 512u; i += nt) {
        wS1[i] = (half_t)Ws1[i];
        wA1[i] = (half_t)Wa1[i];
    }
    // h0[(b*L+l)][k] = hns[l,0,b,k]
    half_t* h0 = ws + HO_H0;
    for (size_t i = tid; i < (size_t)NBL * HID; i += nt) {
        int bl = (int)(i >> 9), k = (int)(i & 511);
        int b = bl >> 7, l = bl & 127;
        h0[i] = (half_t)hns[((size_t)l * BB + b) * HID + k];
    }
}

// ---------------------------------------------------------------- 1) fused 11-layer encoder
// 64 rows / workgroup, double-buffered LDS activations, 2x2 register blocking.
__global__ __launch_bounds__(256) void enc_kernel(
    const float* __restrict__ x, const half_t* __restrict__ wEnc, EncB eb,
    half_t* __restrict__ encOut)
{
    __shared__ __align__(32) half_t sAct[2 * 64 * 192];  // 48 KB
    const int base = blockIdx.x * 64;
    const int tid  = threadIdx.x;
    const int wave = tid >> 5, lane = tid & 31;
    const int mblk = wave & 1;           // 2 m-blocks of 32 rows
    const int nblk = wave >> 1;          // up to 4 n-blocks of 32 cols

    // load + f32->f16 convert the 182-wide features (cols 182..191 zero pad)
    for (int i = tid; i < 64 * 192; i += 256) {
        int m = i / 192, c = i % 192;
        float v = (c < FEAT) ? x[(size_t)(base + m) * (FEAT + 3) + c] : 0.0f;
        sAct[i] = (half_t)v;
    }
    __syncthreads();

    int cur = 0;
    for (int layer = 0; layer < 11; ++layer) {
        const int in_dp  = (layer == 0) ? 192 : 128;
        const int out_d  = (layer < 10) ? 128 : 64;
        const half_t* W  = wEnc + (size_t)layer * 128 * 192;
        const float*  bp = eb.b[layer];
        if (layer < 10) __builtin_prefetch(wEnc + (size_t)(layer + 1) * 128 * 192, 0, 1);

        half_t* src = sAct + cur * (64 * 192);
        half_t* dst = sAct + (cur ^ 1) * (64 * 192);
        if (nblk * 32 < out_d) {                     // wave-uniform predicate
            const int m0 = mblk * 32;
            const int n0 = nblk * 32;
            v8f a00 = {}, a01 = {}, a10 = {}, a11 = {};
            for (int k0 = 0; k0 < in_dp; k0 += 32) {
                v16h A0 = ld_A(src, 192, m0,      k0);
                v16h A1 = ld_A(src, 192, m0 + 16, k0);
                v16h B0 = ld_B(W,   192, n0,      k0);
                v16h B1 = ld_B(W,   192, n0 + 16, k0);
                a00 = wmma16(A0, B0, a00);
                a01 = wmma16(A0, B1, a01);
                a10 = wmma16(A1, B0, a10);
                a11 = wmma16(A1, B1, a11);
            }
            const int nA  = n0 + (lane & 15);
            const int nB  = nA + 16;
            const int mb0 = m0 + ((lane >> 4) << 3);
            const int mb1 = mb0 + 16;
            const float bA = bp[nA], bB = bp[nB];
#pragma unroll
            for (int r = 0; r < 8; ++r) {
                float v00 = a00[r] + bA; v00 = v00 > 0.f ? v00 : 0.f;
                float v01 = a01[r] + bB; v01 = v01 > 0.f ? v01 : 0.f;
                float v10 = a10[r] + bA; v10 = v10 > 0.f ? v10 : 0.f;
                float v11 = a11[r] + bB; v11 = v11 > 0.f ? v11 : 0.f;
                dst[(mb0 + r) * 192 + nA] = (half_t)v00;
                dst[(mb0 + r) * 192 + nB] = (half_t)v01;
                dst[(mb1 + r) * 192 + nA] = (half_t)v10;
                dst[(mb1 + r) * 192 + nB] = (half_t)v11;
            }
        }
        __syncthreads();
        cur ^= 1;
    }
    // final activations: [64][64] valid -> encOut row-major [NROW][64]
    const half_t* fin = sAct + cur * (64 * 192);
    for (int i = tid; i < 64 * 64; i += 256) {
        int m = i >> 6, j = i & 63;
        encOut[(size_t)(base + m) * 64 + j] = fin[m * 192 + j];
    }
}

// ---------------------------------------------------------------- 2) gates + LSTM cell (fused)
// 32 (b,l)-rows per workgroup; enc/h0 tiles async-copied into LDS.
__global__ __launch_bounds__(256) void gates_kernel(
    const half_t* __restrict__ enc, const half_t* __restrict__ h0,
    const half_t* __restrict__ wIh, const half_t* __restrict__ wHh,
    const float* __restrict__ b_ih, const float* __restrict__ b_hh,
    const float* __restrict__ cns, half_t* __restrict__ hnOut,
    float* __restrict__ outHn, float* __restrict__ outC)
{
    __shared__ __align__(32) half_t sE[32 * 512];    // 32 KB
    __shared__ __align__(32) half_t sH[32 * 512];    // 32 KB
    const int m0g  = blockIdx.x * 32;
    const int tid  = threadIdx.x;
    const int wave = tid >> 5, lane = tid & 31;

    {   // async stage: 2 x 32 KB, 2048 16B-chunks each, all lanes active
        const u64 gaE = (u64)(const void*)(enc + (size_t)m0g * 512);
        const u64 gaH = (u64)(const void*)(h0  + (size_t)m0g * 512);
        const u32 laE = (u32)(u64)(const void*)sE;
        const u32 laH = (u32)(u64)(const void*)sH;
        for (int ch = tid; ch < 2048; ch += 256) {
            async_cp16(laE + ch * 16, gaE + (u64)ch * 16);
            async_cp16(laH + ch * 16, gaH + (u64)ch * 16);
        }
        async_wait0();
    }
    __syncthreads();

    const int m0w = (wave & 1) * 16;                 // 2 m-tiles of 16 rows
    for (int n0 = (wave >> 1) * 16; n0 < 512; n0 += 64) {
        v8f aI = {}, aF = {}, aG = {}, aO = {};
        for (int k0 = 0; k0 < 512; k0 += 32) {
            v16h aE = ld_A(sE, 512, m0w, k0);
            v16h aH = ld_A(sH, 512, m0w, k0);
            aI = wmma16(aE, ld_B(wIh, 512,        n0, k0), aI);
            aI = wmma16(aH, ld_B(wHh, 512,        n0, k0), aI);
            aF = wmma16(aE, ld_B(wIh, 512,  512 + n0, k0), aF);
            aF = wmma16(aH, ld_B(wHh, 512,  512 + n0, k0), aF);
            aG = wmma16(aE, ld_B(wIh, 512, 1024 + n0, k0), aG);
            aG = wmma16(aH, ld_B(wHh, 512, 1024 + n0, k0), aG);
            aO = wmma16(aE, ld_B(wIh, 512, 1536 + n0, k0), aO);
            aO = wmma16(aH, ld_B(wHh, 512, 1536 + n0, k0), aO);
        }
        const int n  = n0 + (lane & 15);
        const int mb = m0w + ((lane >> 4) << 3);
        const float bI = b_ih[n]        + b_hh[n];
        const float bF = b_ih[512 + n]  + b_hh[512 + n];
        const float bG = b_ih[1024 + n] + b_hh[1024 + n];
        const float bO = b_ih[1536 + n] + b_hh[1536 + n];
#pragma unroll
        for (int r = 0; r < 8; ++r) {
            const int m = m0g + mb + r;
            const int b = m >> 7, l = m & 127;
            const float c0 = cns[((size_t)l * BB + b) * HID + n];
            const float iv = sigm(aI[r] + bI);
            const float fv = sigm(aF[r] + bF);
            const float gv = tanhf(aG[r] + bG);
            const float ov = sigm(aO[r] + bO);
            const float c  = fv * c0 + iv * gv;
            const float hn = ov * tanhf(c);
            hnOut[(size_t)m * 512 + n] = (half_t)hn;
            if (l == LL - 1) {                       // hn[:,-1], c[:,-1] outputs
                outHn[(size_t)b * 512 + n] = hn;
                outC [(size_t)b * 512 + n] = c;
            }
        }
    }
}

// ---------------------------------------------------------------- 3) dueling heads + mask
__global__ __launch_bounds__(256) void heads_kernel(
    const half_t* __restrict__ hn, const half_t* __restrict__ wS1,
    const half_t* __restrict__ wA1, const float* __restrict__ bs1,
    const float* __restrict__ ba1, const float* __restrict__ Ws2,
    const float* __restrict__ bs2, const float* __restrict__ Wa2,
    const float* __restrict__ ba2, const float* __restrict__ x,
    float* __restrict__ qOut)
{
    __shared__ __align__(32) half_t sIn[16 * 512];       // 16 KB
    __shared__ __align__(32) half_t sT[2][16 * 512];     // 32 KB (s-head, a-head)
    __shared__ float sRed[64 * 4];
    const int m0   = blockIdx.x * 16;
    const int tid  = threadIdx.x;
    const int wave = tid >> 5, lane = tid & 31;

    {   // async stage hn tile (16 KB = 1024 chunks)
        const u64 ga = (u64)(const void*)(hn + (size_t)m0 * 512);
        const u32 la = (u32)(u64)(const void*)sIn;
        for (int ch = tid; ch < 1024; ch += 256)
            async_cp16(la + ch * 16, ga + (u64)ch * 16);
        async_wait0();
    }
    __syncthreads();

    for (int it = 0; it < 2; ++it) {                 // 1x2 n-blocking
        const int n0 = wave * 32 + it * 256;
        v8f s0 = {}, s1 = {}, a0 = {}, a1 = {};
        for (int k0 = 0; k0 < 512; k0 += 32) {
            v16h A = ld_A(sIn, 512, 0, k0);
            s0 = wmma16(A, ld_B(wS1, 512, n0,      k0), s0);
            s1 = wmma16(A, ld_B(wS1, 512, n0 + 16, k0), s1);
            a0 = wmma16(A, ld_B(wA1, 512, n0,      k0), a0);
            a1 = wmma16(A, ld_B(wA1, 512, n0 + 16, k0), a1);
        }
        const int nA = n0 + (lane & 15), nB = nA + 16;
        const int mb = (lane >> 4) << 3;
        const float bSA = bs1[nA], bSB = bs1[nB];
        const float bAA = ba1[nA], bAB = ba1[nB];
#pragma unroll
        for (int r = 0; r < 8; ++r) {
            float v;
            v = s0[r] + bSA; v = v > 0.f ? v : 0.f; sT[0][(mb + r) * 512 + nA] = (half_t)v;
            v = s1[r] + bSB; v = v > 0.f ? v : 0.f; sT[0][(mb + r) * 512 + nB] = (half_t)v;
            v = a0[r] + bAA; v = v > 0.f ? v : 0.f; sT[1][(mb + r) * 512 + nA] = (half_t)v;
            v = a1[r] + bAB; v = v > 0.f ? v : 0.f; sT[1][(mb + r) * 512 + nB] = (half_t)v;
        }
    }
    __syncthreads();

    // 64 dot products (16 rows x {sv, a0, a1, a2}), 4 threads each
    {
        const int d = tid >> 2, sub = tid & 3;           // d < 64
        const int m = d >> 2, j = d & 3;
        const float*  wrow = (j == 0) ? Ws2 : (Wa2 + (size_t)(j - 1) * 512);
        const half_t* trow = (j == 0) ? &sT[0][m * 512] : &sT[1][m * 512];
        float s = 0.0f;
        for (int k = sub * 128; k < sub * 128 + 128; ++k)
            s += (float)trow[k] * wrow[k];
        sRed[d * 4 + sub] = s;
    }
    __syncthreads();

    if (tid < 16) {
        const int m  = tid;
        const int bl = m0 + m;
        float sv = sRed[(m*4+0)*4+0] + sRed[(m*4+0)*4+1] + sRed[(m*4+0)*4+2] + sRed[(m*4+0)*4+3] + bs2[0];
        float av[3];
#pragma unroll
        for (int j = 0; j < 3; ++j)
            av[j] = sRed[(m*4+1+j)*4+0] + sRed[(m*4+1+j)*4+1] +
                    sRed[(m*4+1+j)*4+2] + sRed[(m*4+1+j)*4+3] + ba2[j];
        const float mean = (av[0] + av[1] + av[2]) * (1.0f / 3.0f);
        const float* maskp = x + (size_t)bl * HH * (FEAT + 3) + FEAT;   // x[b,l,0,182..184]
#pragma unroll
        for (int j = 0; j < 3; ++j) {
            float q = sv + (av[j] - mean);
            q += (1.0f - maskp[j]) * -1e33f;
            qOut[(size_t)bl * 3 + j] = q;
        }
    }
}

// ---------------------------------------------------------------- launch
extern "C" void kernel_launch(void* const* d_in, const int* in_sizes, int n_in,
                              void* d_out, int out_size, void* d_ws, size_t ws_size,
                              hipStream_t stream) {
    const float* x    = (const float*)d_in[0];
    const float* hns  = (const float*)d_in[1];
    const float* cns  = (const float*)d_in[2];
    EncW ew; EncB eb;
    for (int i = 0; i < 11; ++i) ew.W[i] = (const float*)d_in[3 + i];
    for (int i = 0; i < 11; ++i) eb.b[i] = (const float*)d_in[14 + i];
    const float* W_ih = (const float*)d_in[25];
    const float* b_ih = (const float*)d_in[26];
    const float* W_hh = (const float*)d_in[27];
    const float* b_hh = (const float*)d_in[28];
    const float* Ws1  = (const float*)d_in[29];
    const float* bs1  = (const float*)d_in[30];
    const float* Ws2  = (const float*)d_in[31];
    const float* bs2  = (const float*)d_in[32];
    const float* Wa1  = (const float*)d_in[33];
    const float* ba1  = (const float*)d_in[34];
    const float* Wa2  = (const float*)d_in[35];
    const float* ba2  = (const float*)d_in[36];

    half_t* ws = (half_t*)d_ws;
    float*  q    = (float*)d_out;            // 256*128*3
    float*  oHn  = q + (size_t)NBL * 3;      // 1*256*512
    float*  oC   = oHn + (size_t)BB * HID;   // 1*256*512

    cvt_kernel<<<8192, 256, 0, stream>>>(ew, W_ih, W_hh, Ws1, Wa1, hns, ws);

    enc_kernel<<<NROW / 64, 256, 0, stream>>>(x, ws + HO_WENC, eb, ws + HO_ENC);

    gates_kernel<<<NBL / 32, 256, 0, stream>>>(
        ws + HO_ENC, ws + HO_H0, ws + HO_WIH, ws + HO_WHH,
        b_ih, b_hh, cns, ws + HO_HN, oHn, oC);

    heads_kernel<<<NBL / 16, 256, 0, stream>>>(
        ws + HO_HN, ws + HO_WS1, ws + HO_WA1, bs1, ba1,
        Ws2, bs2, Wa2, ba2, x, q);
}
